// GRUNet_59115929862247
// MI455X (gfx1250) — matrix operand order repaired and checked by reading, not compiled, not run on hardware
//
#include <hip/hip_runtime.h>

typedef __attribute__((ext_vector_type(16))) __bf16       bf16x16;
typedef __attribute__((ext_vector_type(8)))  float        f32x8;
typedef __attribute__((ext_vector_type(4)))  float        f32x4;
typedef __attribute__((ext_vector_type(4)))  unsigned int u32x4;

union BPack { bf16x16 v; u32x4 q[2]; unsigned short s[16]; };
union QPack { u32x4 q; unsigned short s[8]; };
union SPack { unsigned int u[2]; unsigned short s[4]; };

__device__ __forceinline__ unsigned short f2bf_u(float f) {
    unsigned u = __builtin_bit_cast(unsigned, f);
    u += 0x7FFFu + ((u >> 16) & 1u);            // round-to-nearest-even
    return (unsigned short)(u >> 16);
}
__device__ __forceinline__ float bf2f(unsigned short h) {
    unsigned u = ((unsigned)h) << 16;
    return __builtin_bit_cast(float, u);
}
__device__ __forceinline__ float sigm(float x) { return 1.0f / (1.0f + __expf(-x)); }
__device__ __forceinline__ float tanhfast(float x) {
    float e = __expf(-2.0f * x);
    return (1.0f - e) / (1.0f + e);
}

#define S_DIM 256
#define H_DIM 512
#define A_DIM 32
#define M_ROWS 32          // batch rows per wave (2 M-tiles share every B fetch)
#define WAVES  4           // waves per WGP march in lockstep -> B tiles shared via WGP$

// ws layout (ushort elements): W1b[131072] | W2b[262144] | Wmb[16384] | Whhb[3072]
#define OFF_W2  131072
#define OFF_WM  393216
#define OFF_WHH 409600
#define CVT_TOTAL 412672

__global__ void cvt_bf16_kernel(const float* __restrict__ W1, const float* __restrict__ W2,
                                const float* __restrict__ Wm, const float* __restrict__ whh,
                                unsigned short* __restrict__ o) {
    int i = blockIdx.x * blockDim.x + threadIdx.x;
    if (i < OFF_W2)            o[i] = f2bf_u(W1[i]);
    else if (i < OFF_WM)       o[i] = f2bf_u(W2[i - OFF_W2]);
    else if (i < OFF_WHH)      o[i] = f2bf_u(Wm[i - OFF_WM]);
    else if (i < CVT_TOTAL)    o[i] = f2bf_u(whh[i - OFF_WHH]);
}

// 4 waves per block; each wave privately owns a 32-row strip. Barriers only pace the
// waves so they touch the same weight tile together (WGP$ sharing), never for data.
__global__ __launch_bounds__(32 * WAVES) void grunet_fused(
    const float* __restrict__ x,
    const float* __restrict__ b1,
    const float* __restrict__ b2,
    const float* __restrict__ bmv,
    const float* __restrict__ w_ih,
    const float* __restrict__ b_ih,
    const float* __restrict__ b_hh,
    const float* __restrict__ Ww,
    const float* __restrict__ bw,
    const int*   __restrict__ pred_len,
    const unsigned short* __restrict__ wsb,
    float* __restrict__ out)
{
    const unsigned short* W1b = wsb;
    const unsigned short* W2b = wsb + OFF_W2;
    const unsigned short* Wmb = wsb + OFF_WM;
    const unsigned short* Whb = wsb + OFF_WHH;

    __shared__ unsigned short s_h1a[WAVES * M_ROWS * H_DIM];  // 128KB: h1; later zb+wp per wave
    __shared__ unsigned short s_h2a[WAVES * M_ROWS * H_DIM];  // 128KB: x-stage then h2

    const int lane = threadIdx.x & 31;
    const int wv   = threadIdx.x >> 5;
    const int col  = lane & 15;       // N-column (B/C/D) or M-row-in-tile (A)
    const int hl   = lane >> 4;       // lane-half: K-chunk (A/B) / +8 M-offset (C)
    const size_t rowBase = ((size_t)blockIdx.x * WAVES + wv) * M_ROWS;

    unsigned short* s_h1 = s_h1a + wv * (M_ROWS * H_DIM);
    unsigned short* s_h2 = s_h2a + wv * (M_ROWS * H_DIM);

    // ---------------- Stage x strip -> bf16 in LDS (non-temporal: keep WGP$ for weights) ----
    {
        const f32x4* xs = (const f32x4*)(x + rowBase * S_DIM);
        for (int i = lane; i < (M_ROWS * S_DIM) / 4; i += 32) {
            f32x4 f = __builtin_nontemporal_load(xs + i);
            SPack p;
            #pragma unroll
            for (int e = 0; e < 4; ++e) p.s[e] = f2bf_u(f[e]);
            *(unsigned int*)(s_h2 + i * 4)     = p.u[0];
            *(unsigned int*)(s_h2 + i * 4 + 2) = p.u[1];
        }
    }
    __syncthreads();

    // ---------------- Layer 1: h1 = relu(x @ W1^T + b1) ----------------
    for (int n0 = 0; n0 < H_DIM; n0 += 16) {
        float bias = b1[n0 + col];
        f32x8 acc0 = {bias, bias, bias, bias, bias, bias, bias, bias};
        f32x8 acc1 = acc0;
        #pragma unroll
        for (int k0 = 0; k0 < S_DIM; k0 += 32) {
            BPack wb;                                   // B tile fetched once, feeds 2 WMMAs
            const u32x4* pb = (const u32x4*)(W1b + (size_t)(n0 + col) * S_DIM + k0 + hl * 8);
            wb.q[0] = pb[0]; wb.q[1] = pb[2];
            BPack a0, a1;
            a0.q[0] = *(const u32x4*)(s_h2 + (col)      * S_DIM + k0 + hl * 8);
            a0.q[1] = *(const u32x4*)(s_h2 + (col)      * S_DIM + k0 + 16 + hl * 8);
            a1.q[0] = *(const u32x4*)(s_h2 + (16 + col) * S_DIM + k0 + hl * 8);
            a1.q[1] = *(const u32x4*)(s_h2 + (16 + col) * S_DIM + k0 + 16 + hl * 8);
            acc0 = __builtin_amdgcn_wmma_f32_16x16x32_bf16(false, a0.v, false, wb.v,
                                                           (short)0, acc0, false, false);
            acc1 = __builtin_amdgcn_wmma_f32_16x16x32_bf16(false, a1.v, false, wb.v,
                                                           (short)0, acc1, false, false);
        }
        #pragma unroll
        for (int j = 0; j < 8; ++j) {
            float v0 = acc0[j] > 0.0f ? acc0[j] : 0.0f;
            float v1 = acc1[j] > 0.0f ? acc1[j] : 0.0f;
            s_h1[(j + 8 * hl) * H_DIM + n0 + col]        = f2bf_u(v0);
            s_h1[(16 + j + 8 * hl) * H_DIM + n0 + col]   = f2bf_u(v1);
        }
        __syncthreads();   // pacing: keep waves on the same W1 tile
    }

    // ---------------- Layer 2: h2 = relu(h1 @ W2^T + b2) ----------------
    for (int n0 = 0; n0 < H_DIM; n0 += 16) {
        float bias = b2[n0 + col];
        f32x8 acc0 = {bias, bias, bias, bias, bias, bias, bias, bias};
        f32x8 acc1 = acc0;
        #pragma unroll
        for (int k0 = 0; k0 < H_DIM; k0 += 32) {
            BPack wb;
            const u32x4* pb = (const u32x4*)(W2b + (size_t)(n0 + col) * H_DIM + k0 + hl * 8);
            wb.q[0] = pb[0]; wb.q[1] = pb[2];
            BPack a0, a1;
            a0.q[0] = *(const u32x4*)(s_h1 + (col)      * H_DIM + k0 + hl * 8);
            a0.q[1] = *(const u32x4*)(s_h1 + (col)      * H_DIM + k0 + 16 + hl * 8);
            a1.q[0] = *(const u32x4*)(s_h1 + (16 + col) * H_DIM + k0 + hl * 8);
            a1.q[1] = *(const u32x4*)(s_h1 + (16 + col) * H_DIM + k0 + 16 + hl * 8);
            acc0 = __builtin_amdgcn_wmma_f32_16x16x32_bf16(false, a0.v, false, wb.v,
                                                           (short)0, acc0, false, false);
            acc1 = __builtin_amdgcn_wmma_f32_16x16x32_bf16(false, a1.v, false, wb.v,
                                                           (short)0, acc1, false, false);
        }
        #pragma unroll
        for (int j = 0; j < 8; ++j) {
            float v0 = acc0[j] > 0.0f ? acc0[j] : 0.0f;
            float v1 = acc1[j] > 0.0f ? acc1[j] : 0.0f;
            s_h2[(j + 8 * hl) * H_DIM + n0 + col]        = f2bf_u(v0);   // x-stage is dead now
            s_h2[(16 + j + 8 * hl) * H_DIM + n0 + col]   = f2bf_u(v1);
        }
        __syncthreads();   // pacing: keep waves on the same W2 tile
    }

    // ---------------- z0 = h2 @ Wm^T + bm  (N=32, 2 M-tiles) ----------------
    unsigned short* zb = s_h1;                        // bf16 z-buffer [32][32] (h1 dead)
    f32x8 zreg[2][2];                                 // [m-tile][n-tile]
    #pragma unroll
    for (int t = 0; t < 2; ++t) {
        int n0 = t * 16;
        float bias = bmv[n0 + col];
        f32x8 acc0 = {bias, bias, bias, bias, bias, bias, bias, bias};
        f32x8 acc1 = acc0;
        #pragma unroll
        for (int k0 = 0; k0 < H_DIM; k0 += 32) {
            BPack wb;
            const u32x4* pb = (const u32x4*)(Wmb + (size_t)(n0 + col) * H_DIM + k0 + hl * 8);
            wb.q[0] = pb[0]; wb.q[1] = pb[2];
            BPack a0, a1;
            a0.q[0] = *(const u32x4*)(s_h2 + (col)      * H_DIM + k0 + hl * 8);
            a0.q[1] = *(const u32x4*)(s_h2 + (col)      * H_DIM + k0 + 16 + hl * 8);
            a1.q[0] = *(const u32x4*)(s_h2 + (16 + col) * H_DIM + k0 + hl * 8);
            a1.q[1] = *(const u32x4*)(s_h2 + (16 + col) * H_DIM + k0 + 16 + hl * 8);
            acc0 = __builtin_amdgcn_wmma_f32_16x16x32_bf16(false, a0.v, false, wb.v,
                                                           (short)0, acc0, false, false);
            acc1 = __builtin_amdgcn_wmma_f32_16x16x32_bf16(false, a1.v, false, wb.v,
                                                           (short)0, acc1, false, false);
        }
        zreg[0][t] = acc0; zreg[1][t] = acc1;
        #pragma unroll
        for (int j = 0; j < 8; ++j) {
            zb[(j + 8 * hl) * A_DIM + n0 + col]      = f2bf_u(acc0[j]);
            zb[(16 + j + 8 * hl) * A_DIM + n0 + col] = f2bf_u(acc1[j]);
        }
    }

    // ---------------- GRU loop (recurrent weights live in registers; no barriers) ---------
    BPack wB[6];
    float wih0[6], wih1[6], bihv[6], bhhv[6];
    #pragma unroll
    for (int t = 0; t < 6; ++t) {
        int n = t * 16 + col;
        const u32x4* pb = (const u32x4*)(Whb + n * A_DIM + hl * 8);
        wB[t].q[0] = pb[0]; wB[t].q[1] = pb[2];
        wih0[t] = w_ih[2 * n]; wih1[t] = w_ih[2 * n + 1];
        bihv[t] = b_ih[n];     bhhv[t] = b_hh[n];
    }
    float wwv[32];
    #pragma unroll
    for (int k = 0; k < 32; ++k) wwv[k] = Ww[hl * 32 + k];
    const float bwv = bw[hl];

    // wp broadcast area: floats carved at disjoint offset in this wave's dead s_h1
    float* s_wp = (float*)(s_h1 + 4096);              // 64 floats: [32 rows][2 comps]

    float wp0r[2][8] = {};
    float wp1r[2][8] = {};
    float wpv0 = 0.0f, wpv1 = 0.0f;                   // lane slots: (row=col,hl), (row=16+col,hl)

    const int T = pred_len[0];
    const int ostride = 2 * T;
    float* orow0 = out + (rowBase + (size_t)col) * ostride;
    float* orow1 = orow0 + (size_t)16 * ostride;

    for (int st = 0; st < T; ++st) {
        #pragma unroll
        for (int m = 0; m < 2; ++m) {
            BPack az;
            az.q[0] = *(const u32x4*)(zb + (m * 16 + col) * A_DIM + hl * 8);
            az.q[1] = *(const u32x4*)(zb + (m * 16 + col) * A_DIM + 16 + hl * 8);
            f32x8 gh[6];
            #pragma unroll
            for (int t = 0; t < 6; ++t) {
                float bb = bhhv[t];
                f32x8 c = {bb, bb, bb, bb, bb, bb, bb, bb};
                gh[t] = __builtin_amdgcn_wmma_f32_16x16x32_bf16(false, az.v, false, wB[t].v,
                                                                (short)0, c, false, false);
            }
            #pragma unroll
            for (int t2 = 0; t2 < 2; ++t2) {
                #pragma unroll
                for (int j = 0; j < 8; ++j) {
                    float gr = fmaf(wp0r[m][j], wih0[t2],     fmaf(wp1r[m][j], wih1[t2],     bihv[t2]));
                    float rv = sigm(gh[t2][j] + gr);
                    float gu = fmaf(wp0r[m][j], wih0[2 + t2], fmaf(wp1r[m][j], wih1[2 + t2], bihv[2 + t2]));
                    float uv = sigm(gh[2 + t2][j] + gu);
                    float gn = fmaf(wp0r[m][j], wih0[4 + t2], fmaf(wp1r[m][j], wih1[4 + t2], bihv[4 + t2]));
                    float nv = tanhfast(fmaf(rv, gh[4 + t2][j], gn));
                    float zn = (1.0f - uv) * nv + uv * zreg[m][t2][j];
                    zreg[m][t2][j] = zn;
                    zb[(m * 16 + j + 8 * hl) * A_DIM + t2 * 16 + col] = f2bf_u(zn);
                }
            }
        }
        // wp += z_new @ Ww^T + bw : lane handles rows col and 16+col, component hl
        float dot0 = 0.0f, dot1 = 0.0f;
        #pragma unroll
        for (int c4 = 0; c4 < 4; ++c4) {
            QPack q0, q1;
            q0.q = *(const u32x4*)(zb + (col) * A_DIM + c4 * 8);
            q1.q = *(const u32x4*)(zb + (16 + col) * A_DIM + c4 * 8);
            #pragma unroll
            for (int e = 0; e < 8; ++e) {
                dot0 = fmaf(bf2f(q0.s[e]), wwv[c4 * 8 + e], dot0);
                dot1 = fmaf(bf2f(q1.s[e]), wwv[c4 * 8 + e], dot1);
            }
        }
        wpv0 += dot0 + bwv;
        wpv1 += dot1 + bwv;
        __builtin_nontemporal_store(wpv0, orow0 + st * 2 + hl);
        __builtin_nontemporal_store(wpv1, orow1 + st * 2 + hl);
        s_wp[(col) * 2 + hl]      = wpv0;
        s_wp[(16 + col) * 2 + hl] = wpv1;
        #pragma unroll
        for (int m = 0; m < 2; ++m)
            #pragma unroll
            for (int j = 0; j < 8; ++j) {
                int row = m * 16 + j + 8 * hl;
                wp0r[m][j] = s_wp[row * 2];
                wp1r[m][j] = s_wp[row * 2 + 1];
            }
    }
}

extern "C" void kernel_launch(void* const* d_in, const int* in_sizes, int n_in,
                              void* d_out, int out_size, void* d_ws, size_t ws_size,
                              hipStream_t stream) {
    const float* x    = (const float*)d_in[0];
    const float* W1   = (const float*)d_in[1];
    const float* b1   = (const float*)d_in[2];
    const float* W2   = (const float*)d_in[3];
    const float* b2   = (const float*)d_in[4];
    const float* Wm   = (const float*)d_in[5];
    const float* bmv  = (const float*)d_in[6];
    const float* w_ih = (const float*)d_in[7];
    const float* w_hh = (const float*)d_in[8];
    const float* b_ih = (const float*)d_in[9];
    const float* b_hh = (const float*)d_in[10];
    const float* Ww   = (const float*)d_in[11];
    const float* bw   = (const float*)d_in[12];
    const int*   plen = (const int*)d_in[13];

    unsigned short* wsb = (unsigned short*)d_ws;

    // 1) one-shot bf16 weight conversion into workspace (weights live in L2 thereafter)
    cvt_bf16_kernel<<<(CVT_TOTAL + 255) / 256, 256, 0, stream>>>(W1, W2, Wm, w_hh, wsb);

    // 2) fused MLP + GRU: 4 lockstep waves per WGP, each owning a 32-row strip
    int B = in_sizes[0] / S_DIM;
    grunet_fused<<<B / (M_ROWS * WAVES), 32 * WAVES, 0, stream>>>(
        x, b1, b2, bmv, w_ih, b_ih, b_hh, Ww, bw, plen, wsb, (float*)d_out);
}
